// FusedMoE_30468497997922
// MI455X (gfx1250) — matrix operand order repaired and checked by reading, not compiled, not run on hardware
//
#include <hip/hip_runtime.h>
#include <cstdint>
#include <cstddef>

// ---------------- problem constants (fixed by setup_inputs) ----------------
static constexpr int T  = 2048;
static constexpr int E  = 8;
static constexpr int H  = 1024;
static constexpr int I  = 4096;
static constexpr int N2 = 2 * I;          // 8192 columns of W13
static constexpr int G13 = H / 128;       // 8 scale groups for W13
static constexpr int G2  = I / 128;       // 32 scale groups for W2
static constexpr int LDSK = 40;           // padded LDS stride for 32-wide K tiles (80B, 16B aligned)

typedef __bf16 bf16;
typedef bf16  v16bf __attribute__((ext_vector_type(16)));
typedef bf16  v8bf  __attribute__((ext_vector_type(8)));
typedef float v8f   __attribute__((ext_vector_type(8)));

// ---------------- async global->LDS staging (CDNA5 GLOBAL_LOAD_ASYNC_TO_LDS) ----
#if defined(__gfx1250__) && __has_builtin(__builtin_amdgcn_global_load_async_to_lds_b128) && \
    __has_builtin(__builtin_amdgcn_s_wait_asynccnt)
#define USE_ASYNC_LDS 1
#else
#define USE_ASYNC_LDS 0
#endif

#if USE_ASYNC_LDS
typedef __attribute__((__vector_size__(4 * sizeof(int)))) int v4i_vec;
typedef __attribute__((address_space(1))) v4i_vec* gv4i_p;   // global (AS1)
typedef __attribute__((address_space(3))) v4i_vec* lv4i_p;   // LDS (AS3)
#endif

__device__ __forceinline__ void stage_b128(bf16* lds_dst, const bf16* gsrc) {
#if USE_ASYNC_LDS
  // Generic->AS casts via integers: generic global ptr value == AS1 address;
  // generic LDS ptr low 32 bits == AS3 offset (ISA 10.2 aperture rules).
  __builtin_amdgcn_global_load_async_to_lds_b128(
      (gv4i_p)(uintptr_t)gsrc,
      (lv4i_p)(uint32_t)(uintptr_t)lds_dst, 0, 0);
#else
  *(v8bf*)lds_dst = *(const v8bf*)gsrc;
#endif
}

__device__ __forceinline__ void stage_fence() {
#if USE_ASYNC_LDS
  __builtin_amdgcn_s_wait_asynccnt(0);
#endif
}

// ---------------- fragment loaders (wave32 WMMA 16x16x32 bf16) -------------
// A-matrix 16x32: lane<16 -> M=lane, K = {h8..h8+7, 16+h8..16+h8+7} with h8 = (lane>>4)*8
__device__ __forceinline__ v16bf ldfragA(const bf16* base, int lane) {
  const bf16* p = base + (lane & 15) * LDSK + ((lane >> 4) * 8);
  v8bf lo = *(const v8bf*)(p);
  v8bf hi = *(const v8bf*)(p + 16);
  v16bf r;
#pragma unroll
  for (int i = 0; i < 8; ++i) { r[i] = lo[i]; r[i + 8] = hi[i]; }
  return r;
}

// B-matrix 32x16: lane -> N = lane&15, K = {h8..h8+7, 16+h8..16+h8+7} with h8 = (lane>>4)*8
__device__ __forceinline__ v16bf ldfragB(const bf16* base, int lane) {
  const bf16* p = base + (lane & 15) * LDSK + ((lane >> 4) * 8);
  v8bf lo = *(const v8bf*)(p);
  v8bf hi = *(const v8bf*)(p + 16);
  v16bf r;
#pragma unroll
  for (int i = 0; i < 8; ++i) { r[i] = lo[i]; r[i + 8] = hi[i]; }
  return r;
}

__device__ __forceinline__ v8f vzero8() {
  v8f z = {0.f, 0.f, 0.f, 0.f, 0.f, 0.f, 0.f, 0.f};
  return z;
}

// ---------------- kernel 1: f32 -> bf16 convert of x -----------------------
__global__ void cvt_kernel(const float* __restrict__ x, bf16* __restrict__ xb, int n) {
  int i = blockIdx.x * 256 + threadIdx.x;
  if (i < n) xb[i] = (bf16)x[i];
}

// ---------------- kernel 2: router (softmax -> top2 -> combine wts) --------
__global__ void router_kernel(const float* __restrict__ logits, float* __restrict__ cw) {
  int t = blockIdx.x * 256 + threadIdx.x;
  if (t >= T) return;
  float p[E];
  float mx = -3.4e38f;
#pragma unroll
  for (int e = 0; e < E; ++e) { p[e] = logits[t * E + e]; mx = fmaxf(mx, p[e]); }
#pragma unroll
  for (int e = 0; e < E; ++e) p[e] = __expf(p[e] - mx);   // softmax denom cancels in tw/sum(tw)
  int i1 = 0; float v1 = p[0];
#pragma unroll
  for (int e = 1; e < E; ++e) if (p[e] > v1) { v1 = p[e]; i1 = e; }
  int i2 = -1; float v2 = -1.f;
#pragma unroll
  for (int e = 0; e < E; ++e) if (e != i1 && p[e] > v2) { v2 = p[e]; i2 = e; }
  float inv = __builtin_amdgcn_rcpf(v1 + v2);
#pragma unroll
  for (int e = 0; e < E; ++e) cw[t * E + e] = 0.f;
  cw[t * E + i1] = v1 * inv;
  cw[t * E + i2] = v2 * inv;
}

// ---------------- kernel 3: GEMM1 + SiLU-gate fusion -----------------------
// grid: (I/64, T/128, E); block: 256 threads = 8 waves (4 row-groups x 2 col-groups)
// block tile: 128 tokens x 64 i-columns (computes both gate col i and up col I+i)
__global__ __launch_bounds__(256) void gemm1_kernel(
    const bf16*  __restrict__ xb,      // T x H (bf16)
    const int*   __restrict__ qw,      // E x H/8 x N2 (packed 4b)
    const float* __restrict__ scales,  // E x G13 x N2
    const int*   __restrict__ gidx,    // E x H
    bf16*        __restrict__ act)     // E x T x I (bf16 out)
{
  __shared__ bf16  As[128 * LDSK];      // x tile, row-major, padded
  __shared__ bf16  Bs[128 * LDSK];      // dequant W13 tile, column-per-row layout
  __shared__ int   gl[H];               // g_idx row for this expert
  __shared__ float scl[128 * G13];      // per tile-column: 8 group scales

  const int e   = blockIdx.z;
  const int t0  = blockIdx.y * 128;
  const int i0  = blockIdx.x * 64;
  const int tid = threadIdx.x;
  const int lane = tid & 31;
  const int wave = tid >> 5;
  const int wm = wave & 3;              // 4 row groups of 32 tokens
  const int wn = wave >> 2;             // 2 col groups of 32 i-values

  for (int i = tid; i < H; i += 256) gl[i] = gidx[e * H + i];
  for (int i = tid; i < 128 * G13; i += 256) {
    int c = i >> 3, g = i & 7;
    int n = (c < 64) ? (i0 + c) : (I + i0 + (c - 64));
    scl[i] = scales[((size_t)e * G13 + g) * N2 + n];
  }
  __syncthreads();

  // per-thread constants for the B staging (both reps share kp -> same kg)
  const int kp = tid & 3;
  const int c0 = tid >> 2;              // rep0 column, rep1 column = c0 + 64
  const int n_of_c0 = (c0 < 64) ? (i0 + c0) : (I + i0 + (c0 - 64));
  const int n_of_c1 = I + i0 + c0;      // c1 = c0+64 is always in the "up" half

  v8f acc[2][2][2];                     // [m-tile][gate/up][n-tile]
#pragma unroll
  for (int a = 0; a < 2; ++a)
#pragma unroll
    for (int b = 0; b < 2; ++b)
#pragma unroll
      for (int c = 0; c < 2; ++c) acc[a][b][c] = vzero8();

#pragma unroll 1
  for (int k0 = 0; k0 < H; k0 += 32) {
    const int kg = k0 + kp * 8;
    // ---- stage A: 128 rows x 32 k of x (async copy, no VGPR round-trip) ----
    {
      int row = tid >> 1, half = tid & 1;
      const bf16* src = xb + (size_t)(t0 + row) * H + k0 + half * 16;
      bf16* dst = As + row * LDSK + half * 16;
      stage_b128(dst, src);
      stage_b128(dst + 8, src + 8);
    }
    // ---- stage B: dequant 128 cols x 32 k (512 packed dwords, 2/thread) ----
    int gv[8];
#pragma unroll
    for (int j = 0; j < 8; ++j) gv[j] = gl[kg + j];   // hoisted: shared by both reps
#pragma unroll
    for (int rep = 0; rep < 2; ++rep) {
      int c = c0 + rep * 64;
      int n = rep ? n_of_c1 : n_of_c0;
      unsigned q = (unsigned)qw[((size_t)e * (H / 8) + (kg >> 3)) * N2 + n];
      const float* sc = &scl[c * G13];
      v8bf o;
#pragma unroll
      for (int j = 0; j < 8; ++j) {
        int w = (int)((q >> (4 * j)) & 15u) - 8;
        o[j] = (bf16)((float)w * sc[gv[j]]);
      }
      *(v8bf*)(Bs + c * LDSK + kp * 8) = o;
    }
    // prefetch next K-step's packed weights into L2/L0
    if (k0 + 32 < H)
      __builtin_prefetch(&qw[((size_t)e * (H / 8) + ((kg + 32) >> 3)) * N2 + n_of_c0], 0, 0);
    stage_fence();
    __syncthreads();

    v16bf a[2], bg[2], bu[2];
#pragma unroll
    for (int mt = 0; mt < 2; ++mt) a[mt] = ldfragA(As + (wm * 32 + mt * 16) * LDSK, lane);
#pragma unroll
    for (int nt = 0; nt < 2; ++nt) {
      bg[nt] = ldfragB(Bs + (wn * 32 + nt * 16) * LDSK, lane);
      bu[nt] = ldfragB(Bs + (64 + wn * 32 + nt * 16) * LDSK, lane);
    }
#pragma unroll
    for (int mt = 0; mt < 2; ++mt)
#pragma unroll
      for (int nt = 0; nt < 2; ++nt) {
        acc[mt][0][nt] = __builtin_amdgcn_wmma_f32_16x16x32_bf16(
            false, a[mt], false, bg[nt], (short)0, acc[mt][0][nt], false, false);
        acc[mt][1][nt] = __builtin_amdgcn_wmma_f32_16x16x32_bf16(
            false, a[mt], false, bu[nt], (short)0, acc[mt][1][nt], false, false);
      }
    __syncthreads();
  }

  // ---- epilogue: act = silu(gate) * up, store bf16 (fast rcp, no IEEE div) ----
  const int hi8  = (lane >> 4) * 8;
  const int ncol = lane & 15;
#pragma unroll
  for (int mt = 0; mt < 2; ++mt)
#pragma unroll
    for (int nt = 0; nt < 2; ++nt)
#pragma unroll
      for (int r = 0; r < 8; ++r) {
        float g = acc[mt][0][nt][r];
        float u = acc[mt][1][nt][r];
        float s = g * __builtin_amdgcn_rcpf(1.0f + __expf(-g));
        int t = t0 + wm * 32 + mt * 16 + hi8 + r;
        int i = i0 + wn * 32 + nt * 16 + ncol;
        act[((size_t)e * T + t) * I + i] = (bf16)(s * u);
      }
}

// ---------------- kernel 4: GEMM2 + router-weighted combine ----------------
// grid: (H/128, T/64); block: 256 threads = 8 waves (2 row-groups x 4 col-groups)
__global__ __launch_bounds__(256) void gemm2_kernel(
    const bf16*  __restrict__ act,     // E x T x I (bf16)
    const int*   __restrict__ qw2,     // E x I/8 x H
    const float* __restrict__ sc2,     // E x G2 x H
    const int*   __restrict__ g2,      // E x I
    const float* __restrict__ cw,      // T x E
    float*       __restrict__ out)     // T x H
{
  __shared__ bf16  As[64 * LDSK];
  __shared__ bf16  Bs[128 * LDSK];
  __shared__ int   gl[I];               // 16 KB
  __shared__ float scl[128 * G2];       // 16 KB
  __shared__ float cws[64 * E];         // 2 KB

  const int h0  = blockIdx.x * 128;
  const int t0  = blockIdx.y * 64;
  const int tid = threadIdx.x;
  const int lane = tid & 31;
  const int wave = tid >> 5;
  const int wm = wave & 1;              // 2 row groups of 32 tokens
  const int wn = wave >> 1;             // 4 col groups of 32 h-values

  for (int i = tid; i < 64 * E; i += 256)
    cws[i] = cw[(t0 + (i >> 3)) * E + (i & 7)];

  v8f oacc[2][2];
#pragma unroll
  for (int a = 0; a < 2; ++a)
#pragma unroll
    for (int b = 0; b < 2; ++b) oacc[a][b] = vzero8();

  const int hi8  = (lane >> 4) * 8;
  const int ncol = lane & 15;
  const int kp = tid & 3;
  const int c0 = tid >> 2;              // rep0 column; rep1 column = c0 + 64

#pragma unroll 1
  for (int e = 0; e < E; ++e) {
    for (int i = tid; i < I; i += 256) gl[i] = g2[(size_t)e * I + i];
    for (int i = tid; i < 128 * G2; i += 256) {
      int c = i >> 5, g = i & 31;
      scl[i] = sc2[((size_t)e * G2 + g) * H + h0 + c];
    }
    __syncthreads();

    v8f acc[2][2];
#pragma unroll
    for (int a = 0; a < 2; ++a)
#pragma unroll
      for (int b = 0; b < 2; ++b) acc[a][b] = vzero8();

#pragma unroll 1
    for (int k0 = 0; k0 < I; k0 += 32) {
      const int kg = k0 + kp * 8;
      // ---- stage A: 64 rows x 32 k of act (async, one b128 per thread) ----
      {
        int row = tid >> 2, q = tid & 3;
        stage_b128(As + row * LDSK + q * 8,
                   act + ((size_t)e * T + t0 + row) * I + k0 + q * 8);
      }
      // ---- stage B: dequant 128 cols x 32 k of W2 ----
      int gv[8];
#pragma unroll
      for (int j = 0; j < 8; ++j) gv[j] = gl[kg + j];
#pragma unroll
      for (int rep = 0; rep < 2; ++rep) {
        int c = c0 + rep * 64;
        unsigned q = (unsigned)qw2[((size_t)e * (I / 8) + (kg >> 3)) * H + h0 + c];
        const float* sc = &scl[c * G2];
        v8bf o;
#pragma unroll
        for (int j = 0; j < 8; ++j) {
          int w = (int)((q >> (4 * j)) & 15u) - 8;
          o[j] = (bf16)((float)w * sc[gv[j]]);
        }
        *(v8bf*)(Bs + c * LDSK + kp * 8) = o;
      }
      if (k0 + 32 < I)
        __builtin_prefetch(&qw2[((size_t)e * (I / 8) + ((kg + 32) >> 3)) * H + h0 + c0], 0, 0);
      stage_fence();
      __syncthreads();

      v16bf a[2], b[2];
#pragma unroll
      for (int mt = 0; mt < 2; ++mt) a[mt] = ldfragA(As + (wm * 32 + mt * 16) * LDSK, lane);
#pragma unroll
      for (int nt = 0; nt < 2; ++nt) b[nt] = ldfragB(Bs + (wn * 32 + nt * 16) * LDSK, lane);
#pragma unroll
      for (int mt = 0; mt < 2; ++mt)
#pragma unroll
        for (int nt = 0; nt < 2; ++nt)
          acc[mt][nt] = __builtin_amdgcn_wmma_f32_16x16x32_bf16(
              false, a[mt], false, b[nt], (short)0, acc[mt][nt], false, false);
      __syncthreads();
    }

    // weight by cw[t,e] and accumulate (cws read-only, no race with next e)
#pragma unroll
    for (int mt = 0; mt < 2; ++mt)
#pragma unroll
      for (int nt = 0; nt < 2; ++nt)
#pragma unroll
        for (int r = 0; r < 8; ++r) {
          int row = wm * 32 + mt * 16 + hi8 + r;
          oacc[mt][nt][r] += cws[row * E + e] * acc[mt][nt][r];
        }
  }

#pragma unroll
  for (int mt = 0; mt < 2; ++mt)
#pragma unroll
    for (int nt = 0; nt < 2; ++nt)
#pragma unroll
      for (int r = 0; r < 8; ++r) {
        int t = t0 + wm * 32 + mt * 16 + hi8 + r;
        int h = h0 + wn * 32 + nt * 16 + ncol;
        out[(size_t)t * H + h] = oacc[mt][nt][r];
      }
}

// ---------------- host launcher --------------------------------------------
extern "C" void kernel_launch(void* const* d_in, const int* in_sizes, int n_in,
                              void* d_out, int out_size, void* d_ws, size_t ws_size,
                              hipStream_t stream) {
  (void)in_sizes; (void)n_in; (void)out_size; (void)ws_size;
  const float* x    = (const float*)d_in[0];
  const float* rl   = (const float*)d_in[1];
  const int*   w13q = (const int*)d_in[2];
  const int*   w2q  = (const int*)d_in[3];
  const float* s13  = (const float*)d_in[4];
  const float* s2   = (const float*)d_in[5];
  const int*   g13  = (const int*)d_in[6];
  const int*   g2   = (const int*)d_in[7];
  // d_in[8] = top_k (fixed at 2 in this problem)

  char* ws = (char*)d_ws;
  bf16*  xb  = (bf16*)ws;                                           // T*H*2      = 4 MB
  float* cw  = (float*)(ws + (size_t)T * H * 2);                    // T*E*4      = 64 KB
  bf16*  act = (bf16*)(ws + (size_t)T * H * 2 + (size_t)T * E * 4); // E*T*I*2    = 128 MB
  float* out = (float*)d_out;

  cvt_kernel<<<(T * H + 255) / 256, 256, 0, stream>>>(x, xb, T * H);
  router_kernel<<<(T + 255) / 256, 256, 0, stream>>>(rl, cw);

  dim3 g1(I / 64, T / 128, E);      // 64 x 16 x 8 = 8192 blocks
  gemm1_kernel<<<g1, 256, 0, stream>>>(xb, w13q, s13, g13, act);

  dim3 g2d(H / 128, T / 64);        // 8 x 32 = 256 blocks
  gemm2_kernel<<<g2d, 256, 0, stream>>>(act, w2q, s2, g2, cw, out);
}
